// Sggnn_prepare_test_55465207660970
// MI455X (gfx1250) — matrix operand order repaired and checked by reading, not compiled, not run on hardware
//
#include <hip/hip_runtime.h>
#include <hip/hip_bf16.h>

#define NROW 8192
#define DDIM 512

typedef __attribute__((ext_vector_type(16))) __bf16 v16bf;
typedef __attribute__((ext_vector_type(8)))  float  v8f;

union Frag {
    v16bf v;
    uint4 q[2];
};

// ---------------------------------------------------------------------------
// Async global -> LDS copy of 16 bytes per lane (CDNA5, ASYNCcnt-tracked).
// GVS addressing: SGPR64 base + VGPR 32-bit byte offset.
// ---------------------------------------------------------------------------
__device__ __forceinline__ void async_ld16(unsigned ldsOff, const __bf16* base, int goff) {
    asm volatile("global_load_async_to_lds_b128 %0, %1, %2"
                 :: "v"(ldsOff), "v"(goff), "s"(base) : "memory");
}
__device__ __forceinline__ void wait_async0() {
    asm volatile("s_wait_asynccnt 0" ::: "memory");
}

// ---------------------------------------------------------------------------
// Kernel 1: feature = row_normalize((qf-gf)^2); emit bf16 hi/lo split and
// sq[i] = ||feature_i||^2.  One block (256 thr) per row, 2 cols per thread.
// ---------------------------------------------------------------------------
__global__ __launch_bounds__(256)
void feat_kernel(const float* __restrict__ qf, const float* __restrict__ gf,
                 float* __restrict__ featOut, __bf16* __restrict__ Fhi,
                 __bf16* __restrict__ Flo, float* __restrict__ sqArr) {
    const int row = blockIdx.x;
    const int tid = threadIdx.x;
    __shared__ float red[256];
    const size_t base = (size_t)row * DDIM;

    float x0 = qf[base + tid]       - gf[base + tid];
    float x1 = qf[base + tid + 256] - gf[base + tid + 256];
    float d0 = x0 * x0, d1 = x1 * x1;

    red[tid] = d0 + d1;
    __syncthreads();
    for (int s = 128; s > 0; s >>= 1) {
        if (tid < s) red[tid] += red[tid + s];
        __syncthreads();
    }
    float rs  = red[0];
    float inv = (rs == 0.0f) ? 0.0f : 1.0f / rs;
    __syncthreads();

    float f0 = d0 * inv, f1 = d1 * inv;
    featOut[base + tid]       = f0;
    featOut[base + tid + 256] = f1;

    __bf16 h0 = (__bf16)f0, h1 = (__bf16)f1;
    Fhi[base + tid]       = h0;
    Fhi[base + tid + 256] = h1;
    Flo[base + tid]       = (__bf16)(f0 - (float)h0);
    Flo[base + tid + 256] = (__bf16)(f1 - (float)h1);

    red[tid] = f0 * f0 + f1 * f1;
    __syncthreads();
    for (int s = 128; s > 0; s >>= 1) {
        if (tid < s) red[tid] += red[tid + s];
        __syncthreads();
    }
    if (tid == 0) sqArr[row] = red[0];
}

// ---------------------------------------------------------------------------
// Kernel 2a: zero t[512] and S
// ---------------------------------------------------------------------------
__global__ void zero_kernel(float* __restrict__ t, float* __restrict__ S) {
    int i = blockIdx.x * blockDim.x + threadIdx.x;
    if (i < DDIM) t[i] = 0.0f;
    if (i == 0)   *S = 0.0f;
}

// ---------------------------------------------------------------------------
// Kernel 2b: t[d] = sum_i feature[i][d]  and  S = sum_i sq[i].
// ---------------------------------------------------------------------------
__global__ __launch_bounds__(256)
void colsum_kernel(const float* __restrict__ feat, const float* __restrict__ sqArr,
                   float* __restrict__ t, float* __restrict__ S) {
    const int b = blockIdx.x, tid = threadIdx.x;
    const int r0 = b * 128;
    float a0 = 0.0f, a1 = 0.0f;
    for (int r = 0; r < 128; ++r) {
        size_t base = (size_t)(r0 + r) * DDIM;
        a0 += feat[base + tid];
        a1 += feat[base + tid + 256];
    }
    atomicAdd(&t[tid], a0);
    atomicAdd(&t[tid + 256], a1);

    __shared__ float red[256];
    red[tid] = (tid < 128) ? sqArr[r0 + tid] : 0.0f;
    __syncthreads();
    for (int s = 128; s > 0; s >>= 1) {
        if (tid < s) red[tid] += red[tid + s];
        __syncthreads();
    }
    if (tid == 0) atomicAdd(S, red[0]);
}

// ---------------------------------------------------------------------------
// Kernel 3: analytic row-sum of (adj + I):
//   rowsum_i = N*sq_i + S - 2*(feature_i . t) + 1 ; inv_i = rowsum==0 ? 0 : 1/rowsum
// ---------------------------------------------------------------------------
__global__ __launch_bounds__(256)
void inv_kernel(const float* __restrict__ feat, const float* __restrict__ t,
                const float* __restrict__ sqArr, const float* __restrict__ S,
                float* __restrict__ invArr) {
    const int row = blockIdx.x, tid = threadIdx.x;
    const size_t base = (size_t)row * DDIM;
    float p = feat[base + tid] * t[tid] + feat[base + tid + 256] * t[tid + 256];
    __shared__ float red[256];
    red[tid] = p;
    __syncthreads();
    for (int s = 128; s > 0; s >>= 1) {
        if (tid < s) red[tid] += red[tid + s];
        __syncthreads();
    }
    if (tid == 0) {
        float rowsum = (float)NROW * sqArr[row] + *S - 2.0f * red[0] + 1.0f;
        invArr[row] = (rowsum == 0.0f) ? 0.0f : 1.0f / rowsum;
    }
}

// ---------------------------------------------------------------------------
// Kernel 4: WMMA GEMM, upper-triangular blocks only, fused epilogue + mirror.
//   out[i][j] = (sq_i + sq_j - 2*(F_i . F_j) + (i==j)) * inv_i
//   out[j][i] = (sq_i + sq_j - 2*(F_i . F_j)        ) * inv_j   (bi != bj)
// F.F^T via split-bf16: Ah*Bh + Ah*Bl + Al*Bh (3x v_wmma_f32_16x16x32_bf16).
// Double-buffered async global->LDS staging (ASYNCcnt).
// ---------------------------------------------------------------------------
__global__ __launch_bounds__(256)
void gemm_kernel(const __bf16* __restrict__ Fhi, const __bf16* __restrict__ Flo,
                 const float* __restrict__ sqArr, const float* __restrict__ invArr,
                 float* __restrict__ out) {
    // 2 stages x { Ah | Al | Bh | Bl } x 128x32 bf16 (8KB each) = 64KB,
    // aliased with the 64x129 f32 transpose buffer (33KB) for the mirror pass.
    __shared__ __align__(16) unsigned char smem[2 * 4 * 8192];

    const int tid  = threadIdx.x;
    const int lane = tid & 31;       // wave32
    const int w    = tid >> 5;       // 8 waves
    const int m    = lane & 15;
    const int hf   = lane >> 4;      // lane-half selector

    // Triangular block decode: 2080 blocks -> (bi <= bj)
    int rem = blockIdx.x, bi = 0, len = 64;
    while (rem >= len) { rem -= len; ++bi; --len; }
    const int bj    = bi + rem;
    const int iBase = bi * 128;
    const int jBase = bj * 128;

    const unsigned ldsBase = (unsigned)(size_t)(void*)smem;

    auto stage = [&](int buf, int kk) {
        unsigned sb = ldsBase + (unsigned)buf * 32768u;
        for (int c = tid; c < 512; c += 256) {
            int row = c >> 2, qid = c & 3;
            int gA = (iBase + row) * 1024 + kk * 64 + qid * 16;  // byte offsets
            int gB = (jBase + row) * 1024 + kk * 64 + qid * 16;
            unsigned l = sb + (unsigned)c * 16u;
            async_ld16(l,          Fhi, gA);   // Ah
            async_ld16(l + 8192u,  Flo, gA);   // Al
            async_ld16(l + 16384u, Fhi, gB);   // Bh
            async_ld16(l + 24576u, Flo, gB);   // Bl
        }
    };

    v8f acc[8] = {};

    stage(0, 0);
    for (int kk = 0; kk < DDIM / 32; ++kk) {
        const int cur = kk & 1;
        wait_async0();        // this wave's loads into buf[cur] have landed in LDS
        __syncthreads();      // all waves waited => buf[cur] ready; buf[1-cur] free
        if (kk + 1 < DDIM / 32) stage(1 - cur, kk + 1);

        const unsigned short* sAh = (const unsigned short*)(smem + cur * 32768);
        const unsigned short* sAl = sAh + 4096;
        const unsigned short* sBh = sAh + 8192;
        const unsigned short* sBl = sAh + 12288;

        // A fragment (16-bit 16x32 layout): half 0 -> K 0..7,16..23 ;
        // half 1 -> K 8..15,24..31 ; two 16B runs per lane.
        Frag ah, al;
        {
            const uint4* rH = (const uint4*)(sAh + (w * 16 + m) * 32);
            const uint4* rL = (const uint4*)(sAl + (w * 16 + m) * 32);
            ah.q[0] = rH[hf];  ah.q[1] = rH[2 + hf];
            al.q[0] = rL[hf];  al.q[1] = rL[2 + hf];
        }

        for (int t = 0; t < 8; ++t) {
            // B fragment (32x16): lane = col n; half 0 -> K 0..7,16..23, half 1 -> K 8..15,24..31
            Frag bh, bl;
            const uint4* rH = (const uint4*)(sBh + (t * 16 + m) * 32);
            const uint4* rL = (const uint4*)(sBl + (t * 16 + m) * 32);
            bh.q[0] = rH[hf]; bh.q[1] = rH[2 + hf];
            bl.q[0] = rL[hf]; bl.q[1] = rL[2 + hf];

            acc[t] = __builtin_amdgcn_wmma_f32_16x16x32_bf16(
                false, ah.v, false, bh.v, (short)0, acc[t], false, false);
            acc[t] = __builtin_amdgcn_wmma_f32_16x16x32_bf16(
                false, ah.v, false, bl.v, (short)0, acc[t], false, false);
            acc[t] = __builtin_amdgcn_wmma_f32_16x16x32_bf16(
                false, al.v, false, bh.v, (short)0, acc[t], false, false);
        }
    }

    // -------- direct epilogue: rows i (upper block), coalesced over j --------
    // C/D layout: VGPR r -> M = r + 8*half, N = lane&15.
    float sqi[8], invi[8];
    for (int r = 0; r < 8; ++r) {
        int i = iBase + w * 16 + r + hf * 8;
        sqi[r]  = sqArr[i];
        invi[r] = invArr[i];
    }
    for (int t = 0; t < 8; ++t) {
        int j = jBase + t * 16 + m;
        float sqj = sqArr[j];
        for (int r = 0; r < 8; ++r) {
            int i = iBase + w * 16 + r + hf * 8;
            float val = sqi[r] + sqj - 2.0f * acc[t][r] + ((i == j) ? 1.0f : 0.0f);
            out[(size_t)i * NROW + j] = val * invi[r];
        }
    }

    // -------- mirrored epilogue: rows j (lower block) via LDS transpose ------
    if (bi != bj) {
        float* T = (float*)smem;            // [64][129] floats, conflict-free
        for (int p = 0; p < 2; ++p) {
            __syncthreads();                // staging / previous pass reads done
            for (int tt = 0; tt < 4; ++tt) {
                int t = p * 4 + tt;
                int c = tt * 16 + m;        // local col 0..63
                int r0 = w * 16 + hf * 8;
                for (int r = 0; r < 8; ++r)
                    T[c * 129 + r0 + r] = acc[t][r];
            }
            __syncthreads();
            for (int e = 0; e < 32; ++e) {
                int idx = e * 256 + tid;    // 0..8191
                int c   = idx >> 7;         // 0..63
                int il  = idx & 127;
                int j   = jBase + p * 64 + c;
                int i   = iBase + il;
                float val = sqArr[j] + sqArr[i] - 2.0f * T[c * 129 + il];
                out[(size_t)j * NROW + i] = val * invArr[j];
            }
        }
    }
}

// ---------------------------------------------------------------------------
extern "C" void kernel_launch(void* const* d_in, const int* in_sizes, int n_in,
                              void* d_out, int out_size, void* d_ws, size_t ws_size,
                              hipStream_t stream) {
    const float* qf = (const float*)d_in[0];
    const float* gf = (const float*)d_in[1];

    float* adjOut  = (float*)d_out;                       // [N*N]
    float* featOut = adjOut + (size_t)NROW * NROW;        // [N*D]

    // Workspace layout
    __bf16* Fhi   = (__bf16*)d_ws;                        // N*D bf16 = 8 MB
    __bf16* Flo   = Fhi + (size_t)NROW * DDIM;            // 8 MB
    float*  sqArr = (float*)(Flo + (size_t)NROW * DDIM);  // N floats
    float*  tArr  = sqArr + NROW;                         // D floats
    float*  Sptr  = tArr + DDIM;                          // 1 float (padded)
    float*  invA  = Sptr + 4;                             // N floats

    feat_kernel<<<NROW, 256, 0, stream>>>(qf, gf, featOut, Fhi, Flo, sqArr);
    zero_kernel<<<2, 256, 0, stream>>>(tArr, Sptr);
    colsum_kernel<<<NROW / 128, 256, 0, stream>>>(featOut, sqArr, tArr, Sptr);
    inv_kernel<<<NROW, 256, 0, stream>>>(featOut, tArr, sqArr, Sptr, invA);

    const int nBlocks = (64 * 65) / 2;   // upper-triangular 128x128 blocks
    gemm_kernel<<<nBlocks, 256, 0, stream>>>(Fhi, Flo, sqArr, invA, adjOut);
}
